// GCNModel_52836687675718
// MI455X (gfx1250) — matrix operand order repaired and checked by reading, not compiled, not run on hardware
//
#include <hip/hip_runtime.h>
#include <hip/hip_bf16.h>

// ---------------------------------------------------------------------------
// GCN (2x GCNConv + Linear) for MI455X / gfx1250.
//
// - Dense transforms: v_wmma_f32_16x16x32_bf16, fp32 accumulate.
// - A tiles staged into LDS with GLOBAL_LOAD_ASYNC_TO_LDS_B128 (ASYNCcnt /
//   s_wait_asynccnt), double buffered -> 4x less global A traffic, and the
//   self-loop + bias epilogue is fused into the GEMM (saves a 200MB pass
//   per layer at 23.3 TB/s).
// - Edge aggregation: hardware global_atomic_add_f32; the 100MB H matrix
//   sits in the 192MB L2, so the random gather/scatter stays on-chip.
// - 100000 rows % 32 == 0 -> no WMMA boundary masking (EXEC all-ones req).
// ---------------------------------------------------------------------------

typedef __bf16 v16bf __attribute__((ext_vector_type(16)));
typedef float  v8f   __attribute__((ext_vector_type(8)));
typedef unsigned short v8u16  __attribute__((ext_vector_type(8)));
typedef unsigned short v16u16 __attribute__((ext_vector_type(16)));

union U16x16 { v16u16 u; v16bf b; };

#define N_NODES 100000
#define N_EDGES 800000

// fp32 -> bf16 (round to nearest even), bit-level
__device__ __forceinline__ unsigned short f2bf(float f) {
    unsigned int u = __float_as_uint(f);
    u += 0x7FFFu + ((u >> 16) & 1u);
    return (unsigned short)(u >> 16);
}

// -------------------------- small utility kernels --------------------------

__global__ void k_fill1(float* __restrict__ p, int n) {
    int i = blockIdx.x * blockDim.x + threadIdx.x;
    if (i < n) p[i] = 1.0f;                       // self-loop contributes 1
}

__global__ void k_deg_count(float* __restrict__ deg,
                            const long long* __restrict__ dst, int e) {
    int i = blockIdx.x * blockDim.x + threadIdx.x;
    if (i < e) unsafeAtomicAdd(&deg[(int)dst[i]], 1.0f);
}

__global__ void k_rsqrt(float* __restrict__ p, int n) {
    int i = blockIdx.x * blockDim.x + threadIdx.x;
    if (i < n) p[i] = rsqrtf(p[i]);
}

__global__ void k_cvt_bf16(const float* __restrict__ in,
                           unsigned short* __restrict__ out, int n) {
    int i = blockIdx.x * blockDim.x + threadIdx.x;
    if (i < n) out[i] = f2bf(in[i]);
}

// W[K][N] fp32 -> Wt[N][K] bf16 (weights are tiny; one-time transform)
__global__ void k_transpose_w(const float* __restrict__ W,
                              unsigned short* __restrict__ Wt, int K, int N) {
    int i = blockIdx.x * blockDim.x + threadIdx.x;
    if (i < K * N) {
        int k = i / N, n = i % N;
        Wt[(size_t)n * K + k] = f2bf(W[i]);
    }
}

// one block per edge, one thread per feature dim (D == blockDim.x == 256)
__global__ void k_scatter(float* __restrict__ Out,
                          const float* __restrict__ Hlin,
                          const float* __restrict__ dinv,
                          const long long* __restrict__ src,
                          const long long* __restrict__ dst, int D) {
    int e = blockIdx.x;
    int s = (int)src[e];
    int t = (int)dst[e];
    float w = dinv[s] * dinv[t];
    int d = threadIdx.x;
    unsafeAtomicAdd(&Out[(size_t)t * D + d], Hlin[(size_t)s * D + d] * w);
}

__global__ void k_relu_cvt(const float* __restrict__ in,
                           unsigned short* __restrict__ out, int n) {
    int i = blockIdx.x * blockDim.x + threadIdx.x;
    if (i < n) out[i] = f2bf(fmaxf(in[i], 0.0f));
}

// --------------------- WMMA GEMM with async LDS staging --------------------
// Hlin[M x 256] = A[M x KDIM](bf16) @ Bt[256 x KDIM]^T(bf16)
// Agg[M x 256]  = Hlin * dinv[row]^2 + bias[col]     (fused epilogue)
//
// Block: 256 threads = 8 waves (2 M-waves x 4 N-waves), 32-row tile.
// A tile (32 x 64 K, 4KB) is DMA'd to LDS via GLOBAL_LOAD_ASYNC_TO_LDS_B128
// (one 16B chunk per thread per stage), double buffered. LDS rows padded to
// 72 elements (144B) so the 16 fragment lanes hit distinct bank groups.
template <int KDIM>
__global__ __launch_bounds__(256) void gemm_bf16_wmma_lds(
    const unsigned short* __restrict__ A,
    const unsigned short* __restrict__ Bt,
    float* __restrict__ Hlin,
    float* __restrict__ Agg,
    const float* __restrict__ dinv,
    const float* __restrict__ bias) {
    constexpr int NDIM = 256;
    constexpr int NSTAGE = KDIM / 64;
    constexpr int LROW = 72;                       // padded row stride (elems)
    __shared__ unsigned short tile[2][32 * LROW];  // 2 x 4.5KB

    const int tid  = threadIdx.x;
    const int lane = tid & 31;
    const int wave = tid >> 5;
    const int wm = wave & 1;
    const int wn = wave >> 1;
    const int r    = lane & 15;   // A: row-in-tile, B: col-in-tile, C: col
    const int half = lane >> 4;

    const int rowB = blockIdx.x * 32;
    const int row0 = rowB + wm * 16;
    const int col0 = wn * 64;

    // async-copy geometry: 256 threads x 16B == 32 rows x 128B (64 K-elems)
    const int crow   = tid >> 3;
    const int cchunk = tid & 7;
    const unsigned short* gsrc = A + (size_t)(rowB + crow) * KDIM + cchunk * 8;
    const unsigned ldsoff = (unsigned)(crow * (LROW * 2) + cchunk * 16);
    const unsigned lb0 = (unsigned)(unsigned long long)&tile[0][0];
    const unsigned lb1 = (unsigned)(unsigned long long)&tile[1][0];

    // B: lane loads 16 contiguous K starting at 16*half each k-step
    const unsigned short* bbase[4];
#pragma unroll
    for (int j = 0; j < 4; ++j)
        bbase[j] = Bt + (size_t)(col0 + j * 16 + r) * KDIM + (half << 4);

    v8f acc[4] = {};

    // prologue: stage 0 copy
    {
        unsigned ldsa = lb0 + ldsoff;
        unsigned long long ga = (unsigned long long)gsrc;
        asm volatile("global_load_async_to_lds_b128 %0, %1, off"
                     :: "v"(ldsa), "v"(ga) : "memory");
    }

    for (int s = 0; s < NSTAGE; ++s) {
        if (s + 1 < NSTAGE) {
            unsigned ldsa = (((s + 1) & 1) ? lb1 : lb0) + ldsoff;
            unsigned long long ga =
                (unsigned long long)(gsrc + (s + 1) * 64);
            asm volatile("global_load_async_to_lds_b128 %0, %1, off"
                         :: "v"(ldsa), "v"(ga) : "memory");
            asm volatile("s_wait_asynccnt 0x1" ::: "memory");
        } else {
            asm volatile("s_wait_asynccnt 0x0" ::: "memory");
        }
        __syncthreads();   // all waves' DMA for stage s visible in LDS

        const unsigned short* lb = &tile[s & 1][0];
        // A frag: lane needs K [16*half,+16) per 32-K step
        const unsigned short* ap0 = lb + (wm * 16 + r) * LROW + half * 16;
#pragma unroll
        for (int ks = 0; ks < 64; ks += 32) {
            v8u16 alo = *(const v8u16*)(ap0 + ks);
            v8u16 ahi = *(const v8u16*)(ap0 + ks + 8);
            U16x16 a;
            a.u = __builtin_shufflevector(alo, ahi, 0, 1, 2, 3, 4, 5, 6, 7,
                                          8, 9, 10, 11, 12, 13, 14, 15);
            const int kg = s * 64 + ks;
#pragma unroll
            for (int j = 0; j < 4; ++j) {
                U16x16 b;
                b.u = *(const v16u16*)(bbase[j] + kg);
                acc[j] = __builtin_amdgcn_wmma_f32_16x16x32_bf16(
                    false, a.b, false, b.b, (short)0, acc[j], false, false);
            }
        }
        __syncthreads();   // stage s buffer reusable two stages later
    }

    // fused epilogue: Hlin = acc ; Agg = acc * dinv^2 + bias
    float sc2[8];
#pragma unroll
    for (int g = 0; g < 8; ++g) {
        float sc = dinv[row0 + g + 8 * half];
        sc2[g] = sc * sc;
    }
#pragma unroll
    for (int j = 0; j < 4; ++j) {
        const int col = col0 + j * 16 + r;
        const float bb = bias[col];
#pragma unroll
        for (int g = 0; g < 8; ++g) {
            const int rr = row0 + g + 8 * half;   // C/D layout: M = g + 8*half
            const float v = acc[j][g];
            Hlin[(size_t)rr * NDIM + col] = v;
            Agg[(size_t)rr * NDIM + col]  = v * sc2[g] + bb;
        }
    }
}

// ---------------- direct-load WMMA GEMM (classifier, N=64) -----------------
template <int KDIM, int NDIM, int WM, int WN>
__global__ void gemm_bf16_wmma(const unsigned short* __restrict__ A,
                               const unsigned short* __restrict__ Bt,
                               float* __restrict__ C,
                               const float* __restrict__ bias) {
    const int lane = threadIdx.x & 31;
    const int wave = threadIdx.x >> 5;
    const int wm = wave % WM;
    const int wn = wave / WM;
    const int r    = lane & 15;
    const int half = lane >> 4;

    const int row0 = blockIdx.x * (WM * 16) + wm * 16;
    const int col0 = wn * 64;

    const unsigned short* arow = A + (size_t)(row0 + r) * KDIM + (half << 4);
    const unsigned short* bbase[4];
#pragma unroll
    for (int j = 0; j < 4; ++j)
        bbase[j] = Bt + (size_t)(col0 + j * 16 + r) * KDIM + (half << 4);

    v8f acc[4] = {};

    for (int k0 = 0; k0 < KDIM; k0 += 32) {
        v8u16 alo = *(const v8u16*)(arow + k0);
        v8u16 ahi = *(const v8u16*)(arow + k0 + 8);
        U16x16 a;
        a.u = __builtin_shufflevector(alo, ahi, 0, 1, 2, 3, 4, 5, 6, 7,
                                      8, 9, 10, 11, 12, 13, 14, 15);
#pragma unroll
        for (int j = 0; j < 4; ++j) {
            U16x16 b;
            b.u = *(const v16u16*)(bbase[j] + k0);
            acc[j] = __builtin_amdgcn_wmma_f32_16x16x32_bf16(
                false, a.b, false, b.b, (short)0, acc[j], false, false);
        }
    }

#pragma unroll
    for (int j = 0; j < 4; ++j) {
        const int col = col0 + j * 16 + r;
        const float bb = bias ? bias[col] : 0.0f;
#pragma unroll
        for (int g = 0; g < 8; ++g) {
            const int rr = row0 + g + 8 * half;
            C[(size_t)rr * NDIM + col] = acc[j][g] + bb;
        }
    }
}

// ------------------------------- driver ------------------------------------

extern "C" void kernel_launch(void* const* d_in, const int* in_sizes, int n_in,
                              void* d_out, int out_size, void* d_ws, size_t ws_size,
                              hipStream_t stream) {
    const float*     x   = (const float*)d_in[0];
    const long long* ei  = (const long long*)d_in[1];   // int64 [2, E] flat
    const float*     W1  = (const float*)d_in[2];
    const float*     b1  = (const float*)d_in[3];
    const float*     W2  = (const float*)d_in[4];
    const float*     b2  = (const float*)d_in[5];
    const float*     Wc  = (const float*)d_in[6];
    const float*     bc  = (const float*)d_in[7];
    float* out = (float*)d_out;

    const long long* src = ei;
    const long long* dst = ei + N_EDGES;

    const int N = N_NODES, E = N_EDGES;

    // ---- workspace carve-out (all offsets 256B aligned) ----
    char* w = (char*)d_ws;
    size_t o = 0;
    auto take = [&](size_t bytes) {
        void* p = w + o;
        o += (bytes + 255) & ~(size_t)255;
        return p;
    };
    float*          dinv = (float*)take((size_t)N * 4);            //  0.4 MB
    unsigned short* Wt1  = (unsigned short*)take(512 * 256 * 2);   //  256 KB
    unsigned short* Wt2  = (unsigned short*)take(256 * 256 * 2);   //  128 KB
    unsigned short* Wtc  = (unsigned short*)take(256 * 64 * 2);    //   32 KB
    unsigned short* Xb   = (unsigned short*)take((size_t)N * 512 * 2); // 100 MB (reused as bf16 H)
    float*          Hlin = (float*)take((size_t)N * 256 * 4);      // 100 MB
    float*          Agg  = (float*)take((size_t)N * 256 * 4);      // 100 MB
    (void)ws_size; (void)in_sizes; (void)n_in; (void)out_size;

    const int T = 256;

    // ---- degrees: deg = 1 + in-degree; dinv = rsqrt(deg) ----
    k_fill1<<<(N + T - 1) / T, T, 0, stream>>>(dinv, N);
    k_deg_count<<<(E + T - 1) / T, T, 0, stream>>>(dinv, dst, E);
    k_rsqrt<<<(N + T - 1) / T, T, 0, stream>>>(dinv, N);

    // ---- precision prep ----
    k_cvt_bf16<<<((size_t)N * 512 + T - 1) / T, T, 0, stream>>>(x, Xb, N * 512);
    k_transpose_w<<<(512 * 256 + T - 1) / T, T, 0, stream>>>(W1, Wt1, 512, 256);
    k_transpose_w<<<(256 * 256 + T - 1) / T, T, 0, stream>>>(W2, Wt2, 256, 256);
    k_transpose_w<<<(256 * 64 + T - 1) / T, T, 0, stream>>>(Wc, Wtc, 256, 64);

    const int nElems256 = N * 256;

    // ---- layer 1: GEMM 100000x512x256 (+fused self/bias), scatter, ReLU ----
    gemm_bf16_wmma_lds<512><<<N / 32, 256, 0, stream>>>(Xb, Wt1, Hlin, Agg, dinv, b1);
    k_scatter<<<E, 256, 0, stream>>>(Agg, Hlin, dinv, src, dst, 256);
    unsigned short* Hb = Xb;   // reuse x-bf16 buffer for layer-1 activations
    k_relu_cvt<<<nElems256 / T, T, 0, stream>>>(Agg, Hb, nElems256);

    // ---- layer 2: GEMM 100000x256x256 (+fused self/bias), scatter, ReLU ----
    gemm_bf16_wmma_lds<256><<<N / 32, 256, 0, stream>>>(Hb, Wt2, Hlin, Agg, dinv, b2);
    k_scatter<<<E, 256, 0, stream>>>(Agg, Hlin, dinv, src, dst, 256);
    k_relu_cvt<<<nElems256 / T, T, 0, stream>>>(Agg, Hb, nElems256);

    // ---- classifier: GEMM 100000x256x64 + bias straight to d_out ----
    gemm_bf16_wmma<256, 64, 2, 1><<<N / 32, 64, 0, stream>>>(Hb, Wtc, out, bc);
}
